// DCell_36953898615444
// MI455X (gfx1250) — matrix operand order repaired and checked by reading, not compiled, not run on hardware
//
#include <hip/hip_runtime.h>

#define NSUB 2048
#define BATCH 32
#define MAXI 256
#define MAXO 80
#define AS 264   // LDS row stride (bf16 elems) for A/W tiles (256 + 8 pad)
#define HS 81    // LDS row stride (floats) for h tile
#define NTHREADS 320

typedef __attribute__((ext_vector_type(16))) __bf16 v16bf;
typedef __attribute__((ext_vector_type(2)))  __bf16 v2bf;
typedef __attribute__((ext_vector_type(8)))  float  v8f;

__device__ __forceinline__ unsigned int pack2_bf16(float lo, float hi) {
#if __has_builtin(__builtin_amdgcn_cvt_pk_bf16_f32)
  v2bf p = __builtin_amdgcn_cvt_pk_bf16_f32(lo, hi);
  return __builtin_bit_cast(unsigned int, p);
#else
  // round-to-nearest-even f32 -> bf16, packed (lo in [15:0], hi in [31:16])
  unsigned int a = __float_as_uint(lo);
  unsigned int b = __float_as_uint(hi);
  a += 0x7FFFu + ((a >> 16) & 1u);
  b += 0x7FFFu + ((b >> 16) & 1u);
  return (a >> 16) | (b & 0xFFFF0000u);
#endif
}

__device__ __forceinline__ float fast_tanh(float x) {
#if __has_builtin(__builtin_amdgcn_tanhf)
  return __builtin_amdgcn_tanhf(x);
#elif __has_builtin(__builtin_amdgcn_tanh_f32)
  return __builtin_amdgcn_tanh_f32(x);
#else
  float r;
  asm volatile("v_tanh_f32 %0, %1" : "=v"(r) : "v"(x));
  return r;
#endif
}

__global__ __launch_bounds__(NTHREADS) void dcell_fused_kernel(
    const float* __restrict__ x, const float* __restrict__ W,
    const float* __restrict__ bias, const float* __restrict__ gamma,
    const float* __restrict__ beta, const float* __restrict__ in_mask,
    const float* __restrict__ out_mask, float* __restrict__ out)
{
  __shared__ __align__(16) unsigned short shA[BATCH * AS];  // bf16 x tile [32][264]
  __shared__ __align__(16) unsigned short shW[MAXO  * AS];  // bf16 W tile [80][264]
  __shared__ __align__(16) float shH[BATCH * HS];           // f32 h tile [32][81]
  __shared__ float shScale[MAXO];
  __shared__ float shShift[MAXO];

  const int s    = blockIdx.x;
  const int tid  = threadIdx.x;
  const int lane = tid & 31;
  const int wave = tid >> 5;

  const float* xs = x       + (size_t)s * BATCH * MAXI;
  const float* Ws = W       + (size_t)s * MAXO  * MAXI;
  const float* ms = in_mask + (size_t)s * MAXI;

  // ---- stage x tile (apply in_mask) into LDS as bf16 ----
  for (int i = tid; i < BATCH * (MAXI / 4); i += NTHREADS) {
    const int r  = i >> 6;       // row (batch index), 64 float4 per row
    const int k4 = i & 63;       // float4 index within row
    float4 v = reinterpret_cast<const float4*>(xs)[i];
    float4 m = reinterpret_cast<const float4*>(ms)[k4];
    uint2 p;
    p.x = pack2_bf16(v.x * m.x, v.y * m.y);
    p.y = pack2_bf16(v.z * m.z, v.w * m.w);
    *reinterpret_cast<uint2*>(&shA[r * AS + k4 * 4]) = p;
  }
  // ---- stage W tile into LDS as bf16 ----
  for (int i = tid; i < MAXO * (MAXI / 4); i += NTHREADS) {
    const int r  = i >> 6;       // output-feature row
    const int k4 = i & 63;
    float4 v = reinterpret_cast<const float4*>(Ws)[i];
    uint2 p;
    p.x = pack2_bf16(v.x, v.y);
    p.y = pack2_bf16(v.z, v.w);
    *reinterpret_cast<uint2*>(&shW[r * AS + k4 * 4]) = p;
  }
  __syncthreads();

  // ---- per-wave 16x16 WMMA tile: bt in {0,1} (batch), ot in 0..4 (features) ----
  const int bt  = wave / 5;
  const int ot  = wave % 5;
  const int hl  = lane >> 4;     // half-wave select
  const int l15 = lane & 15;

  const int arow = bt * 16 + l15;   // A: lane holds row M = l15
  const int brow = ot * 16 + l15;   // B: lane holds col N = l15 (row of W)

  union Frag { unsigned int u[8]; v16bf v; };
  Frag Afr[2], Bfr[2];

  auto load_frag = [&](int kc, Frag& Af, Frag& Bf) {
    #pragma unroll
    for (int vv = 0; vv < 8; ++vv) {
      // ISA 7.12.2: 16-bit A 16x32 layout -> packed K pair per VGPR
      const int ka = kc + 16 * (vv >> 2) + 8 * hl + 2 * (vv & 3);
      Af.u[vv] = *reinterpret_cast<const unsigned int*>(&shA[arow * AS + ka]);
      // ISA 7.12.2: 16-bit B 32x16 layout
      const int kb = kc + 16 * (vv >> 2) + 8 * hl + 2 * (vv & 3);
      Bf.u[vv] = *reinterpret_cast<const unsigned int*>(&shW[brow * AS + kb]);
    }
  };

  // software-pipelined K loop: prefetch frag k+1 while WMMA consumes frag k
  load_frag(0, Afr[0], Bfr[0]);
  v8f acc = {0.f, 0.f, 0.f, 0.f, 0.f, 0.f, 0.f, 0.f};
  #pragma unroll
  for (int it = 0; it < MAXI / 32; ++it) {
    if (it < MAXI / 32 - 1)
      load_frag((it + 1) * 32, Afr[(it + 1) & 1], Bfr[(it + 1) & 1]);
    acc = __builtin_amdgcn_wmma_f32_16x16x32_bf16(
        /*neg_a=*/false, Afr[it & 1].v, /*neg_b=*/false, Bfr[it & 1].v,
        /*c_mod=*/(short)0, acc, /*reuse_a=*/false, /*reuse_b=*/false);
  }

  // ---- bias + tanh, park h in LDS (C layout: lane=N, VGPR v -> M = v + 8*hl) ----
  const float bb = bias[(size_t)s * MAXO + ot * 16 + l15];
  #pragma unroll
  for (int vv = 0; vv < 8; ++vv) {
    const int m = bt * 16 + hl * 8 + vv;
    shH[m * HS + ot * 16 + l15] = fast_tanh(acc[vv] + bb);
  }
  __syncthreads();

  // ---- BatchNorm statistics per output feature (biased var, over B=32) ----
  if (tid < MAXO) {
    float sum = 0.f, sq = 0.f;
    #pragma unroll 8
    for (int b = 0; b < BATCH; ++b) {
      const float h = shH[b * HS + tid];
      sum += h; sq += h * h;
    }
    const float mean = sum * (1.0f / BATCH);
    const float var  = sq  * (1.0f / BATCH) - mean * mean;
    const float inv  = rsqrtf(var + 1e-5f);
    const float g    = gamma[(size_t)s * MAXO + tid];
    const float be   = beta [(size_t)s * MAXO + tid];
    const float om   = out_mask[(size_t)s * MAXO + tid];
    shScale[tid] = g * inv * om;
    shShift[tid] = (be - mean * g * inv) * om;
  }
  __syncthreads();

  // ---- normalize + coalesced store: 2560 = 320 * 8 outputs per subsystem ----
  float* outs = out + (size_t)s * BATCH * MAXO;
  #pragma unroll
  for (int j = 0; j < 8; ++j) {
    const int idx = tid + j * NTHREADS;
    const int o = idx % MAXO;
    const int b = idx / MAXO;
    outs[idx] = shH[b * HS + o] * shScale[o] + shShift[o];
  }
}

extern "C" void kernel_launch(void* const* d_in, const int* in_sizes, int n_in,
                              void* d_out, int out_size, void* d_ws, size_t ws_size,
                              hipStream_t stream) {
  const float* x        = (const float*)d_in[0];
  const float* W        = (const float*)d_in[1];
  const float* b        = (const float*)d_in[2];
  const float* gamma    = (const float*)d_in[3];
  const float* beta     = (const float*)d_in[4];
  const float* in_mask  = (const float*)d_in[5];
  const float* out_mask = (const float*)d_in[6];
  float* out = (float*)d_out;
  dcell_fused_kernel<<<NSUB, NTHREADS, 0, stream>>>(
      x, W, b, gamma, beta, in_mask, out_mask, out);
}
